// TransformerLayer_pooling_16698832846977
// MI455X (gfx1250) — matrix operand order repaired and checked by reading, not compiled, not run on hardware
//
#include <hip/hip_runtime.h>
#include <hip/hip_bf16.h>
#include <math.h>

typedef __attribute__((ext_vector_type(16))) _Float16 v16h;
typedef __attribute__((ext_vector_type(8)))  float    v8f;

#define ACT_NONE 0
#define ACT_GELU 1

// ---- gfx1250 async-copy-to-LDS probe (guarded; sync fallback otherwise) ----
#if defined(__has_builtin)
#  if __has_builtin(__builtin_amdgcn_global_load_async_to_lds_b128) && \
      __has_builtin(__builtin_amdgcn_s_wait_asynccnt)
#    define HAVE_ASYNC_LDS 1
#  endif
#endif
#ifndef HAVE_ASYNC_LDS
#  define HAVE_ASYNC_LDS 0
#endif

#if HAVE_ASYNC_LDS
typedef __attribute__((__vector_size__(4 * sizeof(int)))) int v4i;
typedef __attribute__((address_space(1))) v4i gv4i;   // global int4
typedef __attribute__((address_space(3))) v4i lv4i;   // LDS int4
__device__ __forceinline__ void async_copy16(const void* gsrc, void* ldst) {
    __builtin_amdgcn_global_load_async_to_lds_b128((gv4i*)gsrc, (lv4i*)ldst, 0, 0);
}
__device__ __forceinline__ void async_wait0() {
    __builtin_amdgcn_s_wait_asynccnt(0);
}
#endif

__device__ __forceinline__ float gelu_exact(float x) {
    return 0.5f * x * (1.0f + erff(x * 0.70710678118654752f));
}

// ---------------------------------------------------------------------------
// WMMA GEMM: Y[M,N] = act(X[M,K] @ W[K,N] + bias[N]) (+ R[M,N])
// lda==0 broadcasts row 0 (PMA seed queries). Output type templated (f32/f16).
// Block: 256 threads (8 waves), tile 128x128, BK=32.
// Wave grid 2x4: each wave owns 64x32 = 4x2 accumulators -> 8 WMMA / k-step.
// ---------------------------------------------------------------------------
template <int ACT, bool RES, typename OutT>
__global__ __launch_bounds__(256)
void gemm_wmma_kernel(const float* __restrict__ X, int lda,
                      const float* __restrict__ W,
                      const float* __restrict__ bias,
                      const float* __restrict__ R,
                      OutT* __restrict__ Y,
                      int M, int K, int Ntot)
{
    constexpr int BM = 128, BN = 128, BK = 32;
    __shared__ _Float16 As[BM][BK];   // [m][k]
    __shared__ _Float16 Bs[BN][BK];   // W transposed: [n][k]

    const int tid  = threadIdx.x;
    const int lane = tid & 31;
    const int w    = tid >> 5;
    const int m0   = blockIdx.x * BM;
    const int n0   = blockIdx.y * BN;

    const int wm    = w >> 2;              // 0..1 : 64-row band
    const int wn    = w & 3;               // 0..3 : 32-col band
    const int khalf = lane >> 4;
    const int l15   = lane & 15;

    v8f acc[4][2];
    #pragma unroll
    for (int mi = 0; mi < 4; ++mi)
        #pragma unroll
        for (int ni = 0; ni < 2; ++ni)
            acc[mi][ni] = (v8f){};

    const int ar = tid >> 1;               // 0..127
    const int ac = (tid & 1) * 16;         // 0 or 16

    for (int k0 = 0; k0 < K; k0 += BK) {
        // Stage A tile (fp32 -> f16)
        {
            const int  gm = m0 + ar;
            const bool ok = gm < M;
            const float* src = X + (size_t)(ok ? gm : 0) * (size_t)lda + (k0 + ac);
            #pragma unroll
            for (int i = 0; i < 16; ++i)
                As[ar][ac + i] = ok ? (_Float16)src[i] : (_Float16)0.0f;
            if (ok && (k0 + BK) < K) __builtin_prefetch(src + BK, 0, 1);
        }
        // Stage W tile transposed: Bs[n][k] = W[k0+k][n0+n]
        {
            const int gn = n0 + ar;
            #pragma unroll
            for (int i = 0; i < 16; ++i)
                Bs[ar][ac + i] = (_Float16)W[(size_t)(k0 + ac + i) * (size_t)Ntot + gn];
            if ((k0 + BK) < K)
                __builtin_prefetch(&W[(size_t)(k0 + BK + ac) * (size_t)Ntot + gn], 0, 1);
        }
        __syncthreads();

        // Fragments in contiguous k order (k = 16*khalf + s), same for A and B
        v16h af[4], bf[2];
        #pragma unroll
        for (int mi = 0; mi < 4; ++mi) {
            const int mrow = wm * 64 + mi * 16 + l15;
            #pragma unroll
            for (int p = 0; p < 8; ++p) {
                const int ka = khalf * 16 + 2 * p;
                af[mi][2 * p]     = As[mrow][ka];
                af[mi][2 * p + 1] = As[mrow][ka + 1];
            }
        }
        #pragma unroll
        for (int ni = 0; ni < 2; ++ni) {
            const int ncol = wn * 32 + ni * 16 + l15;
            #pragma unroll
            for (int p = 0; p < 8; ++p) {
                const int kb = khalf * 16 + 2 * p;
                bf[ni][2 * p]     = Bs[ncol][kb];
                bf[ni][2 * p + 1] = Bs[ncol][kb + 1];
            }
        }
        #pragma unroll
        for (int mi = 0; mi < 4; ++mi)
            #pragma unroll
            for (int ni = 0; ni < 2; ++ni)
                acc[mi][ni] = __builtin_amdgcn_wmma_f32_16x16x32_f16(
                    false, af[mi], false, bf[ni], (short)0, acc[mi][ni], false, false);
        __syncthreads();
    }

    // Epilogue: C layout m = v + 8*(lane>>4), n = lane&15
    #pragma unroll
    for (int mi = 0; mi < 4; ++mi) {
        #pragma unroll
        for (int ni = 0; ni < 2; ++ni) {
            const int gn = n0 + wn * 32 + ni * 16 + l15;
            #pragma unroll
            for (int v = 0; v < 8; ++v) {
                const int m = m0 + wm * 64 + mi * 16 + v + 8 * khalf;
                if (m < M) {
                    float y = acc[mi][ni][v] + bias[gn];
                    if (ACT == ACT_GELU) y = gelu_exact(y);
                    const size_t idx = (size_t)m * (size_t)Ntot + gn;
                    if (RES) y += R[idx];
                    Y[idx] = (OutT)y;
                }
            }
        }
    }
}

// ---------------------------------------------------------------------------
// Flash attention (per batch*head), 32-query tile per block, 64-key tiles.
// K/V arrive as f16 -> raw async byte copies to LDS (no transpose needed:
// AV B-fragments gather strided from row-major V). ds=48 padded to 64 for
// the QK^T contraction (pad columns zeroed once). scale folded into Q.
// O = Qh + softmax(Q K^T / sqrt(384)) V into the head's 48 columns.
// ---------------------------------------------------------------------------
__global__ __launch_bounds__(256)
void attn_flash_kernel(const float*    __restrict__ Qg,
                       const _Float16* __restrict__ Kf,
                       const _Float16* __restrict__ Vf,
                       float* __restrict__ Og,
                       int nq, int nk, float scale)
{
    constexpr int DHT = 384, DS = 48;
    __shared__ _Float16 Qs[32][64];   // scaled queries, ds-padded
    __shared__ _Float16 Ks[64][64];   // key tile [key][d], cols 48..63 zero
    __shared__ _Float16 Vs[64][48];   // value tile [key][d] row-major
    __shared__ _Float16 Ps[32][64];   // probabilities (f16)
    __shared__ float    Sb[32][64];   // raw scores (f32)
    __shared__ float rowM[32], rowL[32], rowF[32];
    __shared__ float seg[32][8];

    const int tid  = threadIdx.x;
    const int lane = tid & 31;
    const int w    = tid >> 5;
    const int bh   = blockIdx.y;
    const int b    = bh >> 3;
    const int h    = bh & 7;
    const int q0   = blockIdx.x * 32;

    const size_t qbase = ((size_t)b * (size_t)nq) * DHT + (size_t)h * DS;
    const size_t kbase = ((size_t)b * (size_t)nk) * DHT + (size_t)h * DS;

    // Stage query tile once (scaled), zero-pad rows >= nq and d >= 48
    for (int idx = tid; idx < 32 * 64; idx += 256) {
        const int r = idx >> 6, d = idx & 63;
        float v = 0.0f;
        if (d < DS && (q0 + r) < nq)
            v = Qg[qbase + (size_t)(q0 + r) * DHT + d] * scale;
        Qs[r][d] = (_Float16)v;
    }
    // Zero K pad columns once (never rewritten)
    for (int idx = tid; idx < 64 * 16; idx += 256)
        Ks[idx >> 4][48 + (idx & 15)] = (_Float16)0.0f;
    if (tid < 32) { rowM[tid] = -3.0e38f; rowL[tid] = 0.0f; }

    const int khalf = lane >> 4;
    const int l15   = lane & 15;
    // S-phase tile mapping (8 waves): 2 x 4 tiles of 16x16
    const int smt = w >> 2, snt = w & 3;
    const int s_mrow = smt * 16 + l15;
    const int s_ncol = snt * 16 + l15;
    // AV-phase tile mapping (6 waves): 2 x 3 tiles of 16x16
    const int amt = w & 1, ant = w >> 1;
    const int a_mrow = amt * 16 + l15;
    const int a_ncol = ant * 16 + l15;

    v8f acc = {};

    const int nkt = nk / 64;
    for (int kt = 0; kt < nkt; ++kt) {
        const int kk0 = kt * 64;
        __syncthreads();   // previous iteration fully consumed LDS tiles

#if HAVE_ASYNC_LDS
        // 64 rows x 48 halfs = 64 x 6 chunks of 16B, for both K and V
        for (int idx = tid; idx < 64 * 6; idx += 256) {
            const int r = idx / 6, c = idx % 6;
            const size_t go = kbase + (size_t)(kk0 + r) * DHT + c * 8;
            async_copy16(&Kf[go], &Ks[r][c * 8]);
            async_copy16(&Vf[go], &Vs[r][c * 8]);
        }
        async_wait0();
#else
        for (int idx = tid; idx < 64 * 48; idx += 256) {
            const int r = idx / 48, d = idx % 48;
            const size_t go = kbase + (size_t)(kk0 + r) * DHT + d;
            Ks[r][d] = Kf[go];
            Vs[r][d] = Vf[go];
        }
#endif
        if (kt + 1 < nkt) {   // prefetch next tile's rows
            const size_t gn = kbase + (size_t)(kk0 + 64 + (tid & 63)) * DHT;
            __builtin_prefetch(&Kf[gn], 0, 1);
            __builtin_prefetch(&Vf[gn], 0, 1);
        }
        __syncthreads();

        // ---- S = Q K^T (contraction over padded ds = 64, 2 k-steps) ----
        {
            v8f s = {};
            #pragma unroll
            for (int ks = 0; ks < 2; ++ks) {
                v16h a, bb;
                #pragma unroll
                for (int p = 0; p < 8; ++p) {
                    const int ka = ks * 32 + khalf * 16 + 2 * p;
                    a[2 * p]     = Qs[s_mrow][ka];
                    a[2 * p + 1] = Qs[s_mrow][ka + 1];
                    const int kb = ks * 32 + khalf * 16 + 2 * p;
                    bb[2 * p]     = Ks[s_ncol][kb];
                    bb[2 * p + 1] = Ks[s_ncol][kb + 1];
                }
                s = __builtin_amdgcn_wmma_f32_16x16x32_f16(false, a, false, bb, (short)0, s, false, false);
            }
            #pragma unroll
            for (int v = 0; v < 8; ++v)
                Sb[smt * 16 + v + 8 * khalf][s_ncol] = s[v];
        }
        __syncthreads();

        // ---- online softmax over this 64-key tile ----
        {
            const int r = tid >> 3, sgi = tid & 7;
            float mx = Sb[r][sgi * 8];
            #pragma unroll
            for (int j = 1; j < 8; ++j) mx = fmaxf(mx, Sb[r][sgi * 8 + j]);
            seg[r][sgi] = mx;
        }
        __syncthreads();
        if (tid < 32) {
            float tm = seg[tid][0];
            #pragma unroll
            for (int j = 1; j < 8; ++j) tm = fmaxf(tm, seg[tid][j]);
            const float Mn = fmaxf(rowM[tid], tm);
            rowF[tid] = __expf(rowM[tid] - Mn);
            rowM[tid] = Mn;
        }
        __syncthreads();
        {
            const int r = tid >> 3, sgi = tid & 7;
            const float mr = rowM[r];
            float sum = 0.0f;
            #pragma unroll
            for (int j = 0; j < 8; ++j) {
                const int c = sgi * 8 + j;
                const float e = __expf(Sb[r][c] - mr);
                Ps[r][c] = (_Float16)e;
                sum += e;
            }
            seg[r][sgi] = sum;
        }
        __syncthreads();
        if (tid < 32) {
            float s = 0.0f;
            #pragma unroll
            for (int j = 0; j < 8; ++j) s += seg[tid][j];
            rowL[tid] = rowL[tid] * rowF[tid] + s;
        }

        // ---- acc = acc*f + P V (contraction over 64 keys, 2 k-steps) ----
        if (w < 6) {
            #pragma unroll
            for (int v = 0; v < 8; ++v)
                acc[v] *= rowF[amt * 16 + v + 8 * khalf];
            #pragma unroll
            for (int ks = 0; ks < 2; ++ks) {
                v16h a, bb;
                #pragma unroll
                for (int p = 0; p < 8; ++p) {
                    const int ka = ks * 32 + khalf * 16 + 2 * p;
                    a[2 * p]     = Ps[a_mrow][ka];
                    a[2 * p + 1] = Ps[a_mrow][ka + 1];
                    const int kb = ks * 32 + khalf * 16 + 2 * p;
                    bb[2 * p]     = Vs[kb][a_ncol];       // strided gather (row-major V)
                    bb[2 * p + 1] = Vs[kb + 1][a_ncol];
                }
                acc = __builtin_amdgcn_wmma_f32_16x16x32_f16(false, a, false, bb, (short)0, acc, false, false);
            }
        }
    }
    __syncthreads();

    // ---- epilogue: O = Qh + acc / rowL ----
    if (w < 6) {
        const int dcol = h * DS + a_ncol;
        #pragma unroll
        for (int v = 0; v < 8; ++v) {
            const int lm = amt * 16 + v + 8 * khalf;
            const int gq = q0 + lm;
            if (gq < nq) {
                const size_t idx = ((size_t)b * (size_t)nq + gq) * DHT + dcol;
                Og[idx] = Qg[idx] + acc[v] / rowL[lm];
            }
        }
    }
}

// ---------------------------------------------------------------------------
// Host orchestration
// ---------------------------------------------------------------------------
extern "C" void kernel_launch(void* const* d_in, const int* in_sizes, int n_in,
                              void* d_out, int out_size, void* d_ws, size_t ws_size,
                              hipStream_t stream) {
    (void)in_sizes; (void)n_in; (void)out_size; (void)ws_size;

    // Flattened setup_inputs dict:
    //  [0] x [8,2048,128]
    //  [1..40]  sab_params i: base=1+10*i : qw qb kw kb vw vb o1w o1b o2w o2b
    //  [41..50] pma_params
    //  [51] S [1,1,384]
    auto P = [&](int i) { return (const float*)d_in[i]; };
    const float* X = P(0);

    const int B = 8, N = 2048, DH = 384;
    const size_t Rr = (size_t)B * N;        // 16384 activation rows

    float* ws = (float*)d_ws;
    float*    H0  = ws;                      // [R,384] f32
    float*    H1  = H0 + Rr * DH;            // [R,384] f32
    float*    Qf  = H1 + Rr * DH;            // [R,384] f32
    float*    Of  = Qf + Rr * DH;            // [R,384] f32
    float*    Ff  = Of + Rr * DH;            // [R,1024] f32
    _Float16* K16 = (_Float16*)(Ff + Rr * 1024);   // [R,384] f16
    _Float16* V16 = K16 + Rr * DH;                 // [R,384] f16

    const float scale = 0.05103103630798287f; // 1/sqrt(384)

    auto gemm_f32 = [&](const float* Xp, int lda, const float* W, const float* bias,
                        const float* Res, float* Y, int M, int K, int Nn,
                        int act, bool res) {
        dim3 g((unsigned)((M + 127) / 128), (unsigned)(Nn / 128));
        if (act == ACT_GELU)
            gemm_wmma_kernel<ACT_GELU, false, float><<<g, 256, 0, stream>>>(Xp, lda, W, bias, Res, Y, M, K, Nn);
        else if (res)
            gemm_wmma_kernel<ACT_NONE, true,  float><<<g, 256, 0, stream>>>(Xp, lda, W, bias, Res, Y, M, K, Nn);
        else
            gemm_wmma_kernel<ACT_NONE, false, float><<<g, 256, 0, stream>>>(Xp, lda, W, bias, Res, Y, M, K, Nn);
    };
    auto gemm_f16 = [&](const float* Xp, int lda, const float* W, const float* bias,
                        _Float16* Y, int M, int K, int Nn) {
        dim3 g((unsigned)((M + 127) / 128), (unsigned)(Nn / 128));
        gemm_wmma_kernel<ACT_NONE, false, _Float16><<<g, 256, 0, stream>>>(Xp, lda, W, bias, nullptr, Y, M, K, Nn);
    };
    auto attn = [&](const float* Qg, const _Float16* Kg, const _Float16* Vg, float* Og,
                    int nq, int nk) {
        dim3 g((unsigned)((nq + 31) / 32), (unsigned)(B * 8));
        attn_flash_kernel<<<g, 256, 0, stream>>>(Qg, Kg, Vg, Og, nq, nk, scale);
    };

    // ---- 4 SAB blocks (self-attention on the full set) ----
    const float* h = X;
    int din = 128;
    float* cur = H0;
    float* nxt = H1;
    for (int i = 0; i < 4; ++i) {
        const int base = 1 + 10 * i;
        gemm_f32(h, din, P(base + 0), P(base + 1), nullptr, Qf, (int)Rr, din, DH, ACT_NONE, false);
        gemm_f16(h, din, P(base + 2), P(base + 3), K16, (int)Rr, din, DH);
        gemm_f16(h, din, P(base + 4), P(base + 5), V16, (int)Rr, din, DH);
        attn(Qf, K16, V16, Of, N, N);
        gemm_f32(Of, DH,   P(base + 6), P(base + 7), nullptr, Ff, (int)Rr, DH, 1024, ACT_GELU, false);
        gemm_f32(Ff, 1024, P(base + 8), P(base + 9), Of, cur, (int)Rr, 1024, DH, ACT_NONE, true);
        h = cur;
        float* t = cur; cur = nxt; nxt = t;
        din = DH;
    }

    // ---- PMA: 1 seed query per batch attends to the set ----
    const int base = 41;
    const float* Sd = P(51);
    gemm_f32(Sd, 0,  P(base + 0), P(base + 1), nullptr, Qf, B, DH, DH, ACT_NONE, false); // lda=0 broadcast
    gemm_f16(h, DH,  P(base + 2), P(base + 3), K16, (int)Rr, DH, DH);
    gemm_f16(h, DH,  P(base + 4), P(base + 5), V16, (int)Rr, DH, DH);
    attn(Qf, K16, V16, Of, 1, N);
    gemm_f32(Of, DH, P(base + 6), P(base + 7), nullptr, Ff, B, DH, 512, ACT_GELU, false);
    gemm_f32(Ff, 512, P(base + 8), P(base + 9), Of, (float*)d_out, B, 512, DH, ACT_NONE, true);
}
